// GDEE_81071802680057
// MI455X (gfx1250) — matrix operand order repaired and checked by reading, not compiled, not run on hardware
//
#include <hip/hip_runtime.h>
#include <hip/hip_bf16.h>

#define NTOK 512
#define EMB 300
#define WTE 50
#define INDIM 350
#define HH 256
#define G4 1024
#define FDIM 256
#define NCLS 34
#define CPAD 48
#define MTILE 128

typedef __attribute__((ext_vector_type(16))) __bf16 v16bf;
typedef __attribute__((ext_vector_type(8)))  float  v8f;

union AB16 { uint4 q[2]; v16bf v; };

__device__ __forceinline__ float lrelu(float x) { return x > 0.f ? x : 0.01f * x; }
__device__ __forceinline__ float sigm(float x)  { return 1.f / (1.f + __expf(-x)); }

// ---- WMMA fragment loaders (contiguous K order chosen by this code; A and B match) ----
// 16-bit A 16x32 (this code's order, not ISA table order): lanes 0-15 row M=lane, K k0+0..15; lanes 16-31 same row set,
// K k0+16..31; 2 bf16 per VGPR ascending K.
__device__ __forceinline__ v16bf load_a_frag(const __bf16* rowbase, int k0, int khalf) {
  AB16 u;
  u.q[0] = *(const uint4*)(rowbase + k0 + khalf * 16);
  u.q[1] = *(const uint4*)(rowbase + k0 + 8 + khalf * 16);
  return u.v;
}
// 16-bit B 32x16 (KxN): lane n<16 -> col n, K k0+0..15; lane 16+n -> col n, K k0+16..31.
// W2/Wf stored row-major [n_out][k] == B transposed, so a B column is a contiguous LDS row.
__device__ __forceinline__ v16bf load_b_frag(const __bf16* base, int n, int k0, int kgrp) {
  AB16 u;
  const __bf16* p = base + n * FDIM + k0 + kgrp * 16;
  u.q[0] = ((const uint4*)p)[0];
  u.q[1] = ((const uint4*)p)[1];
  return u.v;
}

// ---- K1: embedding gather + concat -> feat [512 x 350] ----
__global__ void embed_kernel(const int* __restrict__ tok, const int* __restrict__ wty,
                             const float* __restrict__ etab, const float* __restrict__ wtab,
                             float* __restrict__ feat) {
  const int n = blockIdx.x, t = threadIdx.x;
  const int tk = tok[n], w = wty[n];
  for (int d = t; d < EMB; d += 128) feat[n * INDIM + d] = etab[tk * EMB + d];
  for (int d = t; d < WTE; d += 128) feat[n * INDIM + EMB + d] = wtab[w * WTE + d];
}

// ---- K2: pre = feat @ Wih.T + (bih+bhh), both directions [512 x 1024] each ----
__global__ void pre_kernel(const float* __restrict__ feat,
                           const float* __restrict__ WihF, const float* __restrict__ bihF,
                           const float* __restrict__ bhhF,
                           const float* __restrict__ WihB, const float* __restrict__ bihB,
                           const float* __restrict__ bhhB,
                           float* __restrict__ preF, float* __restrict__ preB) {
  const int idx = blockIdx.x * blockDim.x + threadIdx.x;  // < 2*512*1024
  const int which = idx >> 19;
  const int rem = idx & ((1 << 19) - 1);
  const int n = rem >> 10, g = rem & 1023;
  const float* Wih = which ? WihB : WihF;
  float acc = which ? (bihB[g] + bhhB[g]) : (bihF[g] + bhhF[g]);
  const float* fr = feat + n * INDIM;
  const float* wr = Wih + g * INDIM;
  for (int d = 0; d < INDIM; ++d) acc += fr[d] * wr[d];
  (which ? preB : preF)[n * G4 + g] = acc;
}

// ---- K3: sequential LSTM scan, one block per direction ----
__global__ __launch_bounds__(1024)
void lstm_kernel(const float* __restrict__ preF, const float* __restrict__ preB,
                 const float* __restrict__ WhhF, const float* __restrict__ WhhB,
                 float* __restrict__ hOut) {
  __shared__ float hPrev[HH], cSt[HH], gl[G4];
  const int dir = blockIdx.x, t = threadIdx.x;
  const float* pre = dir ? preB : preF;
  const float* Whh = dir ? WhhB : WhhF;
  if (t < HH) { hPrev[t] = 0.f; cSt[t] = 0.f; }
  __syncthreads();
  for (int s = 0; s < NTOK; ++s) {
    const int ts = dir ? (NTOK - 1 - s) : s;
    float acc = pre[ts * G4 + t];
    const float* wr = Whh + t * HH;
    for (int d = 0; d < HH; ++d) acc += wr[d] * hPrev[d];
    gl[t] = acc;
    __syncthreads();
    if (t < HH) {
      const float ig = sigm(gl[t]);
      const float fg = sigm(gl[HH + t]);
      const float gg = tanhf(gl[2 * HH + t]);
      const float og = sigm(gl[3 * HH + t]);
      const float c = fg * cSt[t] + ig * gg;
      const float h = og * tanhf(c);
      cSt[t] = c; hPrev[t] = h;
      hOut[ts * (2 * HH) + dir * HH + t] = h;  // concat [fwd | bwd]
    }
    __syncthreads();
  }
}

// ---- K4: A = h @ W1[:, :512].T + b1 ; B = h @ W1[:, 512:].T   (layer-1 decomposition) ----
__global__ void ab_kernel(const float* __restrict__ h, const float* __restrict__ W1,
                          const float* __restrict__ b1,
                          float* __restrict__ Abuf, float* __restrict__ Bbuf) {
  const int idx = blockIdx.x * blockDim.x + threadIdx.x;  // < 2*512*256
  const int which = idx >> 17;
  const int rem = idx & ((1 << 17) - 1);
  const int row = rem >> 8, k = rem & 255;
  const float* hr = h + row * (2 * HH);
  const float* wr = W1 + k * (4 * HH) + which * (2 * HH);
  float acc = which ? 0.f : b1[k];
  for (int d = 0; d < 2 * HH; ++d) acc += hr[d] * wr[d];
  (which ? Bbuf : Abuf)[row * FDIM + k] = acc;
}

// ---- K5: fused pair MLP: o1 = lrelu(A[i]+B[j]) ; lrelu(o1@W2.T+b2) ; @Wf.T+bf ----
__global__ __launch_bounds__(256)
void pair_mlp_kernel(const float* __restrict__ Abuf, const float* __restrict__ Bbuf,
                     const float* __restrict__ W2, const float* __restrict__ b2,
                     const float* __restrict__ Wf, const float* __restrict__ bf,
                     float* __restrict__ out) {
  extern __shared__ char smem[];
  __bf16* sW2 = (__bf16*)smem;                 // 256*256 bf16 = 128 KB
  __bf16* sWf = sW2 + FDIM * FDIM;             // 48*256 bf16  =  24 KB (rows >=34 zero)
  __bf16* sO1 = sWf + CPAD * FDIM;             // 128*256 bf16 =  64 KB
  __bf16* sO2 = sO1 + MTILE * FDIM;            // 128*256 bf16 =  64 KB
  float*  sB2 = (float*)(sO2 + MTILE * FDIM);  // 256 f32
  float*  sBf = sB2 + FDIM;                    // 48 f32

  const int tid = threadIdx.x;
  const int wave = tid >> 5;
  const int lane = tid & 31;
  const int r0 = blockIdx.x * MTILE;

  // stage weights (fp32 global, L2-resident) -> bf16 LDS
  for (int idx = tid; idx < FDIM * FDIM; idx += 256) sW2[idx] = (__bf16)W2[idx];
  for (int idx = tid; idx < CPAD * FDIM; idx += 256) {
    const int row = idx >> 8;
    sWf[idx] = (row < NCLS) ? (__bf16)Wf[idx] : (__bf16)0.f;
  }
  for (int idx = tid; idx < FDIM; idx += 256) sB2[idx] = b2[idx];
  if (tid < CPAD) sBf[tid] = (tid < NCLS) ? bf[tid] : 0.f;

  // phase 1: build o1 tile (b1 already folded into A by ab_kernel)
  for (int idx = tid; idx < MTILE * FDIM; idx += 256) {
    const int m = idx >> 8, k = idx & 255;
    const int r = r0 + m;
    const int i = r >> 9, j = r & 511;
    sO1[idx] = (__bf16)lrelu(Abuf[i * FDIM + k] + Bbuf[j * FDIM + k]);
  }
  __syncthreads();

  const int mrow0 = wave * 16;          // each wave owns a 16-row M strip
  const int lrow  = lane & 15;
  const int khalf = lane >> 4;

  // phase 2: layer2 (M=16/wave, N=256, K=256): A strip is loop-invariant across
  // all 16 N-tiles -> hoist the 8 K-fragments into registers (64 VGPRs) so the
  // inner loop issues only B-fragment loads against the WMMA chain.
  {
    const __bf16* aRow = sO1 + (mrow0 + lrow) * FDIM;
    v16bf aF[8];
#pragma unroll
    for (int kk = 0; kk < 8; ++kk) aF[kk] = load_a_frag(aRow, kk * 32, khalf);

    for (int nt = 0; nt < 16; ++nt) {
      v8f c = {};
      const int n0 = nt * 16;
#pragma unroll
      for (int kk = 0; kk < 8; ++kk) {
        v16bf b = load_b_frag(sW2, n0 + lrow, kk * 32, khalf);
        c = __builtin_amdgcn_wmma_f32_16x16x32_bf16(false, aF[kk], false, b,
                                                    (short)0, c, false, false);
      }
      const float bias = sB2[n0 + lrow];
#pragma unroll
      for (int v = 0; v < 8; ++v) {
        const float x = lrelu(c[v] + bias);
        sO2[(mrow0 + v + 8 * khalf) * FDIM + n0 + lrow] = (__bf16)x;
      }
    }
  }
  __syncthreads();

  // phase 3: final layer (N padded 34->48), same A-hoist, write logits fp32
  {
    const __bf16* aRow = sO2 + (mrow0 + lrow) * FDIM;
    v16bf aF[8];
#pragma unroll
    for (int kk = 0; kk < 8; ++kk) aF[kk] = load_a_frag(aRow, kk * 32, khalf);

    for (int nt = 0; nt < 3; ++nt) {
      v8f c = {};
      const int n0 = nt * 16;
#pragma unroll
      for (int kk = 0; kk < 8; ++kk) {
        v16bf b = load_b_frag(sWf, n0 + lrow, kk * 32, khalf);
        c = __builtin_amdgcn_wmma_f32_16x16x32_bf16(false, aF[kk], false, b,
                                                    (short)0, c, false, false);
      }
      const int col = n0 + lrow;
      if (col < NCLS) {
        const float bias = sBf[col];
#pragma unroll
        for (int v = 0; v < 8; ++v) {
          const int gr = r0 + mrow0 + v + 8 * khalf;
          out[gr * NCLS + col] = c[v] + bias;
        }
      }
    }
  }
}

extern "C" void kernel_launch(void* const* d_in, const int* in_sizes, int n_in,
                              void* d_out, int out_size, void* d_ws, size_t ws_size,
                              hipStream_t stream) {
  (void)in_sizes; (void)n_in; (void)out_size; (void)ws_size;
  const int*   tok  = (const int*)d_in[0];
  const int*   wty  = (const int*)d_in[1];
  const float* etab = (const float*)d_in[2];
  const float* wtab = (const float*)d_in[3];
  const float* WihF = (const float*)d_in[4];
  const float* WhhF = (const float*)d_in[5];
  const float* bihF = (const float*)d_in[6];
  const float* bhhF = (const float*)d_in[7];
  const float* WihB = (const float*)d_in[8];
  const float* WhhB = (const float*)d_in[9];
  const float* bihB = (const float*)d_in[10];
  const float* bhhB = (const float*)d_in[11];
  const float* W1   = (const float*)d_in[12];
  const float* b1   = (const float*)d_in[13];
  const float* W2   = (const float*)d_in[14];
  const float* b2   = (const float*)d_in[15];
  const float* Wf   = (const float*)d_in[16];
  const float* bf   = (const float*)d_in[17];
  float* out = (float*)d_out;

  // workspace layout (fp32): ~7 MB total
  float* ws   = (float*)d_ws;
  float* feat = ws;                        // 512*350
  float* preF = feat + NTOK * INDIM;       // 512*1024
  float* preB = preF + NTOK * G4;          // 512*1024
  float* hbuf = preB + NTOK * G4;          // 512*512
  float* Abuf = hbuf + NTOK * 2 * HH;      // 512*256
  float* Bbuf = Abuf + NTOK * FDIM;        // 512*256

  embed_kernel<<<NTOK, 128, 0, stream>>>(tok, wty, etab, wtab, feat);
  pre_kernel<<<(2 * NTOK * G4) / 256, 256, 0, stream>>>(feat, WihF, bihF, bhhF,
                                                        WihB, bihB, bhhB, preF, preB);
  lstm_kernel<<<2, 1024, 0, stream>>>(preF, preB, WhhF, WhhB, hbuf);
  ab_kernel<<<(2 * NTOK * FDIM) / 256, 256, 0, stream>>>(hbuf, W1, b1, Abuf, Bbuf);

  const int smem_bytes = (FDIM * FDIM + CPAD * FDIM + 2 * MTILE * FDIM) * 2
                       + (FDIM + CPAD) * 4;  // 287,936 B < 320 KB WGP LDS
  (void)hipFuncSetAttribute((const void*)pair_mlp_kernel,
                            hipFuncAttributeMaxDynamicSharedMemorySize, smem_bytes);
  pair_mlp_kernel<<<(NTOK * NTOK) / MTILE, 256, smem_bytes, stream>>>(
      Abuf, Bbuf, W2, b2, Wf, bf, out);
}
